// LOBMambaV3_6674379178504
// MI455X (gfx1250) — compile-verified, matrix operand order checked by reading, not run on hardware
//
#include <hip/hip_runtime.h>

#define B_SZ   4
#define SEQ    2048
#define TROWS  (B_SZ*SEQ)
#define D_INPUT 40
#define D_MODEL 128
#define D_STATE 16
#define D_INNER 256
#define N_HEADS 4
#define D_HEAD  32
#define KLM     100
#define EPS_RMS 1.1920929e-07f
#define EPS_LN  1e-5f
#define A_FLOOR 1e-4f

typedef __attribute__((ext_vector_type(16))) __bf16 v16bf;
typedef __attribute__((ext_vector_type(8)))  __bf16 v8bf;
typedef __attribute__((ext_vector_type(4)))  __bf16 v4bf;
typedef __attribute__((ext_vector_type(8)))  float  v8f;
typedef __attribute__((ext_vector_type(4)))  float  v4f;
typedef __attribute__((ext_vector_type(4)))  int    v4i;

#if defined(__has_builtin)
#  if __has_builtin(__builtin_amdgcn_global_load_async_to_lds_b128) && \
      __has_builtin(__builtin_amdgcn_s_wait_asynccnt)
#    define USE_ASYNC_LDS 1
#  endif
#endif

__device__ __forceinline__ float siluf(float x){ return x / (1.f + __expf(-x)); }
__device__ __forceinline__ float sigmoidf_(float x){ return 1.f / (1.f + __expf(-x)); }
__device__ __forceinline__ float softplusf(float x){ return (x > 20.f) ? x : log1pf(__expf(x)); }

// ---------------- embed + skip: [T,40] @ [40,128] x2, scaled by hawkes weight
__global__ void k_embed(const float* __restrict__ x, const float* __restrict__ hw,
                        const float* __restrict__ embW, const float* __restrict__ embB,
                        const float* __restrict__ skW,  const float* __restrict__ skB,
                        float* __restrict__ h, float* __restrict__ skip){
  long row = blockIdx.x; int col = threadIdx.x;
  __shared__ float xr[D_INPUT];
  if (col < D_INPUT) xr[col] = x[row*D_INPUT + col];
  __syncthreads();
  float a = embB[col], s = skB[col];
  #pragma unroll
  for (int k = 0; k < D_INPUT; ++k){
    float xv = xr[k];
    a = fmaf(xv, embW[k*D_MODEL+col], a);
    s = fmaf(xv, skW[k*D_MODEL+col], s);
  }
  float w = hw[row];
  h[row*D_MODEL+col]    = a*w;
  skip[row*D_MODEL+col] = s*w;
}

// ---------------- LayerNorm over 128 (one row per block, 128 threads)
__device__ __forceinline__ void ln_row_body(const float* __restrict__ in, float* __restrict__ out,
                                            const float* __restrict__ w, const float* __restrict__ b,
                                            long rowIn, long rowOut){
  __shared__ float red[D_MODEL];
  int t = threadIdx.x;
  float v = in[rowIn*D_MODEL + t];
  red[t] = v; __syncthreads();
  for (int o = D_MODEL/2; o > 0; o >>= 1){ if (t < o) red[t] += red[t+o]; __syncthreads(); }
  float m = red[0] * (1.f/D_MODEL);
  __syncthreads();
  float d = v - m;
  red[t] = d*d; __syncthreads();
  for (int o = D_MODEL/2; o > 0; o >>= 1){ if (t < o) red[t] += red[t+o]; __syncthreads(); }
  float var = red[0] * (1.f/D_MODEL);
  out[rowOut*D_MODEL + t] = d * rsqrtf(var + EPS_LN) * w[t] + b[t];
}
__global__ void k_ln_all(const float* __restrict__ in, float* __restrict__ out,
                         const float* __restrict__ w, const float* __restrict__ b){
  ln_row_body(in, out, w, b, (long)blockIdx.x, (long)blockIdx.x);
}
__global__ void k_ln_last(const float* __restrict__ in, float* __restrict__ out,
                          const float* __restrict__ w, const float* __restrict__ b){
  ln_row_body(in, out, w, b, (long)blockIdx.x*SEQ + (SEQ-1), (long)blockIdx.x);
}

// ---------------- fp32-accurate GEMM via bf16 WMMA 3-product split
// C[T, LDW-wide] = A[T,KDIM] @ W[KDIM,LDW]; block tile 32(M) x 128(N), 8 waves.
// Compile-time LDA/LDW/KDIM: strided staging offsets fold into immediates and
// the K loop fully unrolls. Double-buffered LDS pipeline: next chunk's global
// loads issue before this chunk's WMMAs; one barrier per chunk.
// MODE 0: out0 = silu(C[:, :256]), out1 = C[:, 256:]   (in_proj)
// MODE 1: out0[row*128+col] += C                        (out_proj residual)
#define GTM 32
#define GTN 128
#define LPAD 40
template<int LDA, int LDW, int KDIM, int MODE>
__global__ void __launch_bounds__(256)
k_gemm_bf16x3(const float* __restrict__ A, const float* __restrict__ W,
              float* __restrict__ out0, float* __restrict__ out1){
  constexpr int NC = KDIM / 32;
  __shared__ __bf16 aHi[2][GTM][LPAD],  aLo[2][GTM][LPAD];
  __shared__ __bf16 wHiT[2][GTN][LPAD], wLoT[2][GTN][LPAD];
  int tid  = threadIdx.x;
  int lane = tid & 31, wave = tid >> 5;
  int mwave = wave >> 2, nwave = wave & 3;
  int half = lane >> 4, l16 = lane & 15;
  long rowBase = (long)blockIdx.x * GTM;
  int  ncolBase = blockIdx.y * GTN;
  // staging assignments
  int ar_s = tid >> 3, ak_s = (tid & 7) << 2;   // A: one float4 per thread
  int wn_s = tid >> 1, wk_s = (tid & 1) << 4;   // W: 16 strided loads, contiguous LDS store
  int ar = mwave*16 + l16;
  const float* aPtr = &A[(rowBase + ar_s)*LDA + ak_s];
  const float* wPtr = &W[(long)wk_s*LDW + ncolBase + wn_s];

  auto cvtStore = [&](int buf, v4f av, const float* wv){
    v4bf h4, l4;
    #pragma unroll
    for (int i = 0; i < 4; ++i){
      __bf16 hi = (__bf16)av[i];
      h4[i] = hi; l4[i] = (__bf16)(av[i] - (float)hi);
    }
    *reinterpret_cast<v4bf*>(&aHi[buf][ar_s][ak_s]) = h4;
    *reinterpret_cast<v4bf*>(&aLo[buf][ar_s][ak_s]) = l4;
    v8bf h8a, l8a, h8b, l8b;
    #pragma unroll
    for (int i = 0; i < 8; ++i){
      __bf16 h0 = (__bf16)wv[i];     h8a[i] = h0; l8a[i] = (__bf16)(wv[i]   - (float)h0);
      __bf16 h1 = (__bf16)wv[8+i];   h8b[i] = h1; l8b[i] = (__bf16)(wv[8+i] - (float)h1);
    }
    *reinterpret_cast<v8bf*>(&wHiT[buf][wn_s][wk_s])     = h8a;
    *reinterpret_cast<v8bf*>(&wHiT[buf][wn_s][wk_s + 8]) = h8b;
    *reinterpret_cast<v8bf*>(&wLoT[buf][wn_s][wk_s])     = l8a;
    *reinterpret_cast<v8bf*>(&wLoT[buf][wn_s][wk_s + 8]) = l8b;
  };

  v8f acc0 = {}; v8f acc1 = {};
  // prologue: stage chunk 0
  {
    v4f av = *reinterpret_cast<const v4f*>(aPtr);
    float wv[16];
    #pragma unroll
    for (int i = 0; i < 16; ++i) wv[i] = wPtr[i*LDW];
    cvtStore(0, av, wv);
  }
  __syncthreads();
  #pragma unroll
  for (int kc = 0; kc < NC; ++kc){
    const int cur = kc & 1;
    // issue next chunk's global loads before this chunk's WMMAs
    v4f aN = {};
    float wN[16];
    if (kc + 1 < NC){
      aN = *reinterpret_cast<const v4f*>(aPtr + (kc + 1)*32);
      #pragma unroll
      for (int i = 0; i < 16; ++i) wN[i] = wPtr[(long)(kc + 1)*32*LDW + i*LDW];
    }
    // ---- fragments: two contiguous 16B chunks per lane ----
    v8bf aH0 = *reinterpret_cast<const v8bf*>(&aHi[cur][ar][half*8]);
    v8bf aH1 = *reinterpret_cast<const v8bf*>(&aHi[cur][ar][16 + half*8]);
    v8bf aL0 = *reinterpret_cast<const v8bf*>(&aLo[cur][ar][half*8]);
    v8bf aL1 = *reinterpret_cast<const v8bf*>(&aLo[cur][ar][16 + half*8]);
    v16bf aH = __builtin_shufflevector(aH0, aH1, 0,1,2,3,4,5,6,7,8,9,10,11,12,13,14,15);
    v16bf aL = __builtin_shufflevector(aL0, aL1, 0,1,2,3,4,5,6,7,8,9,10,11,12,13,14,15);
    #pragma unroll
    for (int ns = 0; ns < 2; ++ns){
      int nb = nwave*32 + ns*16 + l16;
      v8bf bH0 = *reinterpret_cast<const v8bf*>(&wHiT[cur][nb][half*8]);
      v8bf bH1 = *reinterpret_cast<const v8bf*>(&wHiT[cur][nb][16 + half*8]);
      v8bf bL0 = *reinterpret_cast<const v8bf*>(&wLoT[cur][nb][half*8]);
      v8bf bL1 = *reinterpret_cast<const v8bf*>(&wLoT[cur][nb][16 + half*8]);
      v16bf bH = __builtin_shufflevector(bH0, bH1, 0,1,2,3,4,5,6,7,8,9,10,11,12,13,14,15);
      v16bf bL = __builtin_shufflevector(bL0, bL1, 0,1,2,3,4,5,6,7,8,9,10,11,12,13,14,15);
      v8f& acc = ns ? acc1 : acc0;
      acc = __builtin_amdgcn_wmma_f32_16x16x32_bf16(false, aH, false, bH, (short)0, acc, false, false);
      acc = __builtin_amdgcn_wmma_f32_16x16x32_bf16(false, aH, false, bL, (short)0, acc, false, false);
      acc = __builtin_amdgcn_wmma_f32_16x16x32_bf16(false, aL, false, bH, (short)0, acc, false, false);
    }
    if (kc + 1 < NC){
      cvtStore(cur ^ 1, aN, wN);   // other buffer: no race with laggard readers of `cur`
      __syncthreads();
    }
  }
  #pragma unroll
  for (int ns = 0; ns < 2; ++ns){
    v8f acc = ns ? acc1 : acc0;
    #pragma unroll
    for (int v = 0; v < 8; ++v){
      long row = rowBase + mwave*16 + (half << 3) + v;   // C layout: VGPR v -> M = v (+8 hi half)
      int gcol = ncolBase + nwave*32 + ns*16 + l16;
      float val = acc[v];
      if (MODE == 0){
        if (gcol < D_INNER) out0[row*D_INNER + gcol] = siluf(val);
        else                out1[row*D_INNER + gcol - D_INNER] = val;
      } else {
        out0[row*D_MODEL + gcol] += val;
      }
    }
  }
}

// ---------------- x_proj (K=34 cols), dt/lam, RMS-norm of B/C (one row / block)
__global__ void __launch_bounds__(256)
k_xproj(const float* __restrict__ xp, const float* __restrict__ Wx,
        const float* __restrict__ Wdt, const float* __restrict__ bdt,
        const float* __restrict__ Bb,  const float* __restrict__ Cb,
        const float* __restrict__ Bnw, const float* __restrict__ Cnw,
        float* __restrict__ dtbuf, float* __restrict__ meandt,
        float* __restrict__ lam, float* __restrict__ Bm, float* __restrict__ Cm){
  long row = blockIdx.x; int t = threadIdx.x;
  __shared__ float xr[D_INNER];
  __shared__ float sp[34];
  __shared__ float red[D_INNER];
  __shared__ float rB, rC;
  xr[t] = xp[row*D_INNER + t];
  __syncthreads();
  if (t < 34){
    float s = 0.f;
    for (int k = 0; k < D_INNER; ++k) s = fmaf(xr[k], Wx[k*34 + t], s);
    sp[t] = s;
  }
  __syncthreads();
  float dtc = softplusf(sp[32]*Wdt[t] + bdt[t]);
  dtbuf[row*D_INNER + t] = dtc;
  red[t] = dtc; __syncthreads();
  for (int o = D_INNER/2; o > 0; o >>= 1){ if (t < o) red[t] += red[t+o]; __syncthreads(); }
  if (t == 0){ meandt[row] = red[0]*(1.f/D_INNER); lam[row] = sigmoidf_(sp[33]); }
  if (t == 1){ float s=0.f; for (int i=0;i<D_STATE;++i){ float v=sp[i]+Bb[i]; s+=v*v; } rB = rsqrtf(s*(1.f/D_STATE)+EPS_RMS); }
  if (t == 2){ float s=0.f; for (int i=0;i<D_STATE;++i){ float v=sp[16+i]+Cb[i]; s+=v*v; } rC = rsqrtf(s*(1.f/D_STATE)+EPS_RMS); }
  __syncthreads();
  if (t < D_STATE)        Bm[row*D_STATE + t]        = (sp[t]    + Bb[t])      * rB * Bnw[t];
  else if (t < 2*D_STATE){ int i = t-D_STATE; Cm[row*D_STATE + i] = (sp[16+i] + Cb[i]) * rC * Cnw[i]; }
}

// ---------------- cumsum of mean(dt): wave32 inclusive scan, 64 chunks of 32
__global__ void k_cumsum(const float* __restrict__ meandt, float* __restrict__ ang){
  int b = blockIdx.x, l = threadIdx.x;   // 32 threads (one wave)
  float run = 0.f;
  for (int t0 = 0; t0 < SEQ; t0 += 32){
    float v = meandt[(long)b*SEQ + t0 + l];
    #pragma unroll
    for (int d = 1; d < 32; d <<= 1){
      float u = __shfl_up(v, d, 32);
      if (l >= d) v += u;
    }
    ang[(long)b*SEQ + t0 + l] = run + v;
    run += __shfl(v, 31, 32);
  }
}

// ---------------- RoPE-style rotation of Bm/Cm (8 pairs each; 16 threads/row)
__global__ void k_rot(const float* __restrict__ ang, const float* __restrict__ rf,
                      float* __restrict__ Bm, float* __restrict__ Cm){
  long row = blockIdx.x; int t = threadIdx.x;
  if (t >= 16) return;
  float* M = (t < 8) ? Bm : Cm;
  int i = t & 7;
  float fr = softplusf(rf[i]);
  float a = ang[row]*fr;
  float ca = cosf(a), sa = sinf(a);
  float e = M[row*D_STATE + 2*i], o = M[row*D_STATE + 2*i + 1];
  M[row*D_STATE + 2*i]     = e*ca - o*sa;
  M[row*D_STATE + 2*i + 1] = e*sa + o*ca;
}

// ---------------- sequential SSM scan; 16-state per (b,c) lane in registers.
// B/C rows staged into double-buffered LDS; next step prefetched with
// global_load_async_to_lds_b128 (ASYNCcnt) when the toolchain exposes it.
__global__ void __launch_bounds__(256)
k_scan(const float* __restrict__ Alog, const float* __restrict__ Bm, const float* __restrict__ Cm,
       const float* __restrict__ dtbuf, const float* __restrict__ lam,
       const float* __restrict__ xp, const float* __restrict__ z,
       const float* __restrict__ Dp, float* __restrict__ y){
  int b = blockIdx.x, c = threadIdx.x;
  float Ac[D_STATE], hst[D_STATE], bxp[D_STATE];
  #pragma unroll
  for (int s = 0; s < D_STATE; ++s){
    Ac[s]  = fminf(-__expf(Alog[c*D_STATE + s]), -A_FLOOR);
    hst[s] = 0.f; bxp[s] = 0.f;
  }
  float Dv = Dp[c];
  __shared__ float sBC[2][2*D_STATE];
#ifdef USE_ASYNC_LDS
  if (c < 8){
    const float* src = ((c < 4) ? Bm : Cm) + ((long)b*SEQ)*D_STATE + (c & 3)*4;
    __builtin_amdgcn_global_load_async_to_lds_b128(
        (v4i*)src, (v4i*)&sBC[0][(c < 4 ? 0 : D_STATE) + (c & 3)*4], 0, 0);
  }
#endif
  for (int t = 0; t < SEQ; ++t){
    long row = (long)b*SEQ + t;
    int pb = t & 1;
#ifdef USE_ASYNC_LDS
    __builtin_amdgcn_s_wait_asynccnt(0);
    __syncthreads();
    if (c < 8 && t + 1 < SEQ){
      const float* src = ((c < 4) ? Bm : Cm) + (row + 1)*D_STATE + (c & 3)*4;
      __builtin_amdgcn_global_load_async_to_lds_b128(
          (v4i*)src, (v4i*)&sBC[pb ^ 1][(c < 4 ? 0 : D_STATE) + (c & 3)*4], 0, 0);
    }
#else
    if (c < D_STATE)        sBC[pb][c] = Bm[row*D_STATE + c];
    else if (c < 2*D_STATE) sBC[pb][c] = Cm[row*D_STATE + c - D_STATE];
    __syncthreads();
#endif
    const float* sB = &sBC[pb][0];
    const float* sC = &sBC[pb][D_STATE];
    float dtv  = dtbuf[row*D_INNER + c];
    float xpv  = xp[row*D_INNER + c];
    float lamv = lam[row];
    float c0 = (1.f - lamv)*dtv, c1 = lamv*dtv;
    float yv = 0.f;
    #pragma unroll
    for (int s = 0; s < D_STATE; ++s){
      float dA = __expf(Ac[s]*dtv);
      float bx = sB[s]*xpv;
      float u  = (t == 0) ? dtv*bx : (c0*dA*bxp[s] + c1*bx);
      hst[s] = fmaf(dA, hst[s], u);
      yv = fmaf(hst[s], sC[s], yv);
      bxp[s] = bx;
    }
    float zv = z[row*D_INNER + c];
    y[row*D_INNER + c] = (yv + xpv*Dv)*siluf(zv);
  }
}

// ---------------- top-100 (rank counting, stable tie-break), sorted ascending
__global__ void __launch_bounds__(256)
k_topk(const float* __restrict__ hw, int* __restrict__ idx){
  int b = blockIdx.x, t = threadIdx.x;
  __shared__ float wv[SEQ];
  __shared__ unsigned char sel[SEQ];
  for (int i = t; i < SEQ; i += 256) wv[i] = hw[(long)b*SEQ + i];
  __syncthreads();
  for (int i = t; i < SEQ; i += 256){
    float vi = wv[i]; int cnt = 0;
    for (int j = 0; j < SEQ; ++j){
      float vj = wv[j];
      cnt += (vj > vi) || (vj == vi && j < i);
    }
    sel[i] = (cnt < KLM) ? 1 : 0;
  }
  __syncthreads();
  for (int i = t; i < SEQ; i += 256){
    if (sel[i]){
      int p = 0;
      for (int j = 0; j < i; ++j) p += sel[j];
      idx[b*KLM + p] = i;
    }
  }
}

// ---------------- landmark attention: 1 query x 100 keys, 4 heads of 32
__global__ void __launch_bounds__(128)
k_attn(const float* __restrict__ moL, const float* __restrict__ skip, const int* __restrict__ idx,
       const float* __restrict__ WQ, const float* __restrict__ bQ,
       const float* __restrict__ WK, const float* __restrict__ bK,
       const float* __restrict__ WV, const float* __restrict__ bV,
       const float* __restrict__ WO, const float* __restrict__ bO,
       float* __restrict__ outp){
  int b = blockIdx.x, t = threadIdx.x;
  __shared__ float kvs[KLM][D_MODEL];   // 51.2 KB (WGP LDS budget is 320 KB)
  __shared__ float q[D_MODEL];
  __shared__ float probs[N_HEADS][KLM];
  __shared__ float ao[D_MODEL];
  for (int e = t; e < KLM*D_MODEL; e += 128){
    int kk = e >> 7, d = e & 127;
    kvs[kk][d] = skip[((long)b*SEQ + idx[b*KLM + kk])*D_MODEL + d];
  }
  {
    float s = bQ[t];
    for (int k = 0; k < D_MODEL; ++k) s = fmaf(moL[b*D_MODEL + k], WQ[k*D_MODEL + t], s);
    q[t] = s;
  }
  __syncthreads();
  for (int e = t; e < N_HEADS*KLM; e += 128){
    int hh = e / KLM, kk = e % KLM;
    float s = 0.f;
    for (int d = 0; d < D_HEAD; ++d){
      int col = hh*D_HEAD + d;
      float kval = bK[col];
      for (int m = 0; m < D_MODEL; ++m) kval = fmaf(kvs[kk][m], WK[m*D_MODEL + col], kval);
      s = fmaf(q[col], kval, s);
    }
    probs[hh][kk] = s * 0.17677669529663687f;  // 1/sqrt(32)
  }
  __syncthreads();
  if (t < N_HEADS){
    float mx = -1e30f;
    for (int kk = 0; kk < KLM; ++kk) mx = fmaxf(mx, probs[t][kk]);
    float sum = 0.f;
    for (int kk = 0; kk < KLM; ++kk){ float e = __expf(probs[t][kk]-mx); probs[t][kk] = e; sum += e; }
    float inv = 1.f/sum;
    for (int kk = 0; kk < KLM; ++kk) probs[t][kk] *= inv;
  }
  __syncthreads();
  {
    int hh = t >> 5;
    float o = 0.f;
    for (int kk = 0; kk < KLM; ++kk){
      float vval = bV[t];
      for (int m = 0; m < D_MODEL; ++m) vval = fmaf(kvs[kk][m], WV[m*D_MODEL + t], vval);
      o = fmaf(probs[hh][kk], vval, o);
    }
    ao[t] = o;
  }
  __syncthreads();
  {
    float s = bO[t];
    for (int k = 0; k < D_MODEL; ++k) s = fmaf(ao[k], WO[k*D_MODEL + t], s);
    outp[b*D_MODEL + t] = s;
  }
}

extern "C" void kernel_launch(void* const* d_in, const int* in_sizes, int n_in,
                              void* d_out, int out_size, void* d_ws, size_t ws_size,
                              hipStream_t stream){
  const float* x    = (const float*)d_in[0];
  const float* hw   = (const float*)d_in[1];
  const float* embW = (const float*)d_in[2];
  const float* embB = (const float*)d_in[3];
  const float* skW  = (const float*)d_in[4];
  const float* skB  = (const float*)d_in[5];
  const float* inW  = (const float*)d_in[6];
  const float* xW   = (const float*)d_in[7];
  const float* dtW  = (const float*)d_in[8];
  const float* dtB  = (const float*)d_in[9];
  const float* Alog = (const float*)d_in[10];
  const float* Bb   = (const float*)d_in[11];
  const float* Cb   = (const float*)d_in[12];
  const float* Bnw  = (const float*)d_in[13];
  const float* Cnw  = (const float*)d_in[14];
  const float* rf   = (const float*)d_in[15];
  const float* Dp   = (const float*)d_in[16];
  const float* outW = (const float*)d_in[17];
  const float* lnw  = (const float*)d_in[18];
  const float* lnb  = (const float*)d_in[19];
  const float* fnw  = (const float*)d_in[20];
  const float* fnb  = (const float*)d_in[21];
  const float* WQ   = (const float*)d_in[22];
  const float* bQ   = (const float*)d_in[23];
  const float* WK   = (const float*)d_in[24];
  const float* bK   = (const float*)d_in[25];
  const float* WV   = (const float*)d_in[26];
  const float* bV   = (const float*)d_in[27];
  const float* WO   = (const float*)d_in[28];
  const float* bO   = (const float*)d_in[29];
  (void)in_sizes; (void)n_in; (void)out_size; (void)ws_size;

  const long T = TROWS;
  float* ws   = (float*)d_ws;
  float* h    = ws; ws += T*D_MODEL;
  float* skip = ws; ws += T*D_MODEL;
  float* xin  = ws; ws += T*D_MODEL;
  float* xp   = ws; ws += T*D_INNER;
  float* z    = ws; ws += T*D_INNER;
  float* dt   = ws; ws += T*D_INNER;
  float* ybuf = ws; ws += T*D_INNER;
  float* Bm   = ws; ws += T*D_STATE;
  float* Cm   = ws; ws += T*D_STATE;
  float* lam  = ws; ws += T;
  float* mdt  = ws; ws += T;
  float* ang  = ws; ws += T;
  float* moL  = ws; ws += B_SZ*D_MODEL;
  int*   idx  = (int*)ws;                    // ~47 MB total workspace

  k_embed<<<(unsigned)T, D_MODEL, 0, stream>>>(x, hw, embW, embB, skW, skB, h, skip);

  for (int L = 0; L < 4; ++L){
    k_ln_all<<<(unsigned)T, D_MODEL, 0, stream>>>(h, xin, lnw + L*D_MODEL, lnb + L*D_MODEL);

    dim3 g1((unsigned)(T/GTM), (2*D_INNER)/GTN);   // 256 x 4 blocks
    k_gemm_bf16x3<D_MODEL, 2*D_INNER, D_MODEL, 0><<<g1, 256, 0, stream>>>(
        xin, inW + (long)L*D_MODEL*2*D_INNER, xp, z);

    k_xproj<<<(unsigned)T, 256, 0, stream>>>(xp, xW + (long)L*D_INNER*34,
                                             dtW + (long)L*D_INNER, dtB + (long)L*D_INNER,
                                             Bb + L*D_STATE, Cb + L*D_STATE,
                                             Bnw + L*D_STATE, Cnw + L*D_STATE,
                                             dt, mdt, lam, Bm, Cm);
    k_cumsum<<<B_SZ, 32, 0, stream>>>(mdt, ang);
    k_rot<<<(unsigned)T, 32, 0, stream>>>(ang, rf + L*8, Bm, Cm);
    k_scan<<<B_SZ, 256, 0, stream>>>(Alog + (long)L*D_INNER*D_STATE, Bm, Cm,
                                     dt, lam, xp, z, Dp + (long)L*D_INNER, ybuf);

    dim3 g2((unsigned)(T/GTM), 1);
    k_gemm_bf16x3<D_INNER, D_MODEL, D_INNER, 1><<<g2, 256, 0, stream>>>(
        ybuf, outW + (long)L*D_INNER*D_MODEL, h, nullptr);
  }

  k_ln_last<<<B_SZ, D_MODEL, 0, stream>>>(h, moL, fnw, fnb);
  k_topk<<<B_SZ, 256, 0, stream>>>(hw, idx);
  k_attn<<<B_SZ, D_MODEL, 0, stream>>>(moL, skip, idx, WQ, bQ, WK, bK, WV, bV, WO, bO,
                                       (float*)d_out);
}
